// WindowAttention_996432413345
// MI455X (gfx1250) — matrix operand order repaired and checked by reading, not compiled, not run on hardware
//
#include <hip/hip_runtime.h>

typedef __attribute__((ext_vector_type(16))) __bf16 v16bf;
typedef __attribute__((ext_vector_type(8)))  float  v8f;
typedef __attribute__((ext_vector_type(4)))  int    v4i;

#define WIN  64
#define REC  128
#define NH   16
#define HD   64
#define DIM  1024
#define NTOK 65
#define MTOK 129
#define BNW  512
#define NEG_BIG (-3.402823466e38f)

union Frag32B { uint4 q[2]; v16bf v; };

// 16 contiguous bf16 (32B, 16B-aligned chunks)
__device__ __forceinline__ v16bf ld_frag32B(const __bf16* p) {
  Frag32B f;
  f.q[0] = *(const uint4*)(p);
  f.q[1] = *(const uint4*)(p + 8);
  return f.v;
}
// A-fragment: elems 0..15 = K[2*kbase..2*kbase+15] (contiguous)
__device__ __forceinline__ v16bf ld_fragA(const __bf16* rowp, int kbase) {
  Frag32B f;
  f.q[0] = *(const uint4*)(rowp + 2 * kbase);
  f.q[1] = *(const uint4*)(rowp + 2 * kbase + 8);
  return f.v;
}

// ---- 16B global->LDS copy: async (CDNA5 GLOBAL_LOAD_ASYNC_TO_LDS_B128) if available ----
#if defined(__has_builtin)
#if __has_builtin(__builtin_amdgcn_global_load_async_to_lds_b128) && \
    __has_builtin(__builtin_amdgcn_s_wait_asynccnt)
#define USE_ASYNC_LDS 1
#endif
#endif

#ifdef USE_ASYNC_LDS
typedef __attribute__((address_space(1))) v4i* as1_v4i;
typedef __attribute__((address_space(3))) v4i* as3_v4i;
#endif

__device__ __forceinline__ void cp16B(const void* g, void* l) {
#ifdef USE_ASYNC_LDS
  // generic->AS1: same 64-bit value; generic LDS ptr low 32 bits == LDS byte address
  __builtin_amdgcn_global_load_async_to_lds_b128(
      (as1_v4i)(unsigned long long)g,
      (as3_v4i)(unsigned int)(unsigned long long)l, 0, 0);
#else
  *(uint4*)l = *(const uint4*)g;
#endif
}
__device__ __forceinline__ void wait_async_all() {
#ifdef USE_ASYNC_LDS
  __builtin_amdgcn_s_wait_asynccnt(0);
#endif
}

// ---------------- fp32 -> bf16 conversion ----------------
__global__ __launch_bounds__(256) void cvt_f32_bf16(const float* __restrict__ src,
                                                    __bf16* __restrict__ dst,
                                                    long long n) {
  long long i = ((long long)blockIdx.x * 256 + threadIdx.x) * 4;
  if (i + 3 < n) {
    const float4 v = *(const float4*)(src + i);
    union { __bf16 h[4]; unsigned long long u; } o;
    o.h[0] = (__bf16)v.x; o.h[1] = (__bf16)v.y;
    o.h[2] = (__bf16)v.z; o.h[3] = (__bf16)v.w;
    *(unsigned long long*)(dst + i) = o.u;
  } else {
    for (long long k = i; k < n; k++) dst[k] = (__bf16)src[k];
  }
}

// ---------------- fused bias+mask table: [cls(5)][h(16)][row(65)][col(160)] ----------------
// cls: 0 -> w==0 (mask_left[0]), 1 -> w==1 (mask_left[1]),
//      2 -> w==nW-2 (mask_right[0]), 3 -> w==nW-1 (mask_right[1]), 4 -> interior (no mask)
__global__ __launch_bounds__(256) void build_bias(
    const float* __restrict__ rel_table, const float* __restrict__ cls_self,
    const float* __restrict__ cls_up, const float* __restrict__ cls_down,
    const unsigned char* __restrict__ mask_left,
    const unsigned char* __restrict__ mask_right,
    float* __restrict__ table) {
  int idx = blockIdx.x * 256 + threadIdx.x;
  const int total = 5 * NH * NTOK * 160;
  if (idx >= total) return;
  int col = idx % 160;
  int row = (idx / 160) % NTOK;
  int h   = (idx / (160 * NTOK)) % NH;
  int cls = idx / (160 * NTOK * NH);
  float v;
  if (col >= MTOK) {
    v = NEG_BIG;
  } else {
    if (row == 0)
      v = (col == 0) ? cls_self[h] : cls_up[h * REC + (col - 1)];
    else
      v = (col == 0) ? cls_down[h * WIN + (row - 1)]
                     : rel_table[(row - col + 127) * NH + h];
    bool m = false;
    if (cls == 0) m = mask_left[((size_t)0 * NTOK + row) * MTOK + col] != 0;
    else if (cls == 1) m = mask_left[((size_t)1 * NTOK + row) * MTOK + col] != 0;
    else if (cls == 2) m = mask_right[((size_t)0 * NTOK + row) * MTOK + col] != 0;
    else if (cls == 3) m = mask_right[((size_t)1 * NTOK + row) * MTOK + col] != 0;
    if (m) v = NEG_BIG;
  }
  table[idx] = v;
}

// ---------------- tiled bf16 WMMA GEMM: C = A(bf16) @ B(bf16) + bias ----------------
// block tile 128(M) x 128(N), K-step 32, double-buffered LDS with async-to-LDS A tiles.
// 8 waves: 4(M) x 2(N); each wave 32x64 -> 8 WMMA per K-step.
// Requires M%128==0, N%128==0, K%32==0.
template <typename OutT>
__global__ __launch_bounds__(256) void gemm_bf16(
    const __bf16* __restrict__ A, const __bf16* __restrict__ B,
    const float* __restrict__ bias, OutT* __restrict__ C,
    int M, int N, int K) {
  __shared__ __bf16 sA[2][128 * 40];   // 128 rows x 32 K, stride 40
  __shared__ __bf16 sBt[2][128 * 48];  // transposed: 128 N x 32 K, stride 48

  const int tid  = threadIdx.x;
  const int wave = tid >> 5, lane = tid & 31;
  const int wr = wave & 3, wc = wave >> 2;
  const int colL = lane & 15, half = lane >> 4;
  const int kbase8 = half * 8, kb16 = half * 16, rbase = half * 8;
  const size_t m0 = (size_t)blockIdx.x * 128;
  const size_t n0 = (size_t)blockIdx.y * 128;
  const int bsegS = tid & 15, browS = tid >> 4;

  v8f acc[2][4] = {};

  auto stageA = [&](int k0, int buf) {
#pragma unroll
    for (int t = 0; t < 2; t++) {
      int idx = tid + t * 256;
      int row = idx >> 2, seg = (idx & 3) * 8;
      cp16B(A + (m0 + row) * (size_t)K + k0 + seg, sA[buf] + row * 40 + seg);
    }
  };
  auto loadB = [&](int k0, uint4* regs) {
#pragma unroll
    for (int rr = 0; rr < 2; rr++)
      regs[rr] = *(const uint4*)(B + (size_t)(k0 + browS + rr * 16) * N + n0 + bsegS * 8);
  };
  auto storeB = [&](const uint4* regs, int buf) {
#pragma unroll
    for (int rr = 0; rr < 2; rr++) {
      int row = browS + rr * 16;
      const __bf16* e = (const __bf16*)&regs[rr];
#pragma unroll
      for (int i = 0; i < 8; i++) sBt[buf][(bsegS * 8 + i) * 48 + row] = e[i];
    }
  };

  // prologue: stage K-tile 0 into buffer 0
  stageA(0, 0);
  {
    uint4 r0[2];
    loadB(0, r0);
    storeB(r0, 0);
  }
  wait_async_all();
  __syncthreads();

  const int nsteps = K / 32;
  for (int s = 0; s < nsteps; s++) {
    const int cur = s & 1, nxt = cur ^ 1;
    const bool more = (s + 1) < nsteps;
    uint4 bregs[2];
    if (more) {                       // overlap next tile's loads with this tile's math
      stageA((s + 1) * 32, nxt);      // async global -> LDS
      loadB((s + 1) * 32, bregs);     // global -> VGPR
    }

    v16bf a0 = ld_fragA(sA[cur] + (wr * 32 + colL) * 40, kbase8);
    v16bf a1 = ld_fragA(sA[cur] + (wr * 32 + 16 + colL) * 40, kbase8);
#pragma unroll
    for (int ns = 0; ns < 4; ns++) {
      v16bf bb = ld_frag32B(sBt[cur] + (wc * 64 + ns * 16 + colL) * 48 + kb16);
      acc[0][ns] = __builtin_amdgcn_wmma_f32_16x16x32_bf16(
          false, a0, false, bb, (short)0, acc[0][ns], false, false);
      acc[1][ns] = __builtin_amdgcn_wmma_f32_16x16x32_bf16(
          false, a1, false, bb, (short)0, acc[1][ns], false, false);
    }

    if (more) storeB(bregs, nxt);     // VGPR -> LDS (other buffer)
    wait_async_all();
    __syncthreads();
  }

#pragma unroll
  for (int mf = 0; mf < 2; mf++) {
#pragma unroll
    for (int ns = 0; ns < 4; ns++) {
#pragma unroll
      for (int r = 0; r < 8; r++) {
        size_t row = m0 + wr * 32 + mf * 16 + rbase + r;
        size_t col = n0 + wc * 64 + ns * 16 + colL;
        C[row * (size_t)N + col] = (OutT)(acc[mf][ns][r] + bias[col]);
      }
    }
  }
}

// ---------------- fused window attention (one block per (window b, head h)) ----------------
__global__ __launch_bounds__(256) void attn_kernel(
    const __bf16* __restrict__ Q, const __bf16* __restrict__ KV,
    const float* __restrict__ biasTable, const int* __restrict__ nW_ptr,
    __bf16* __restrict__ O) {
  extern __shared__ char smem[];
  __bf16* sQ    = (__bf16*)smem;             // [80][72]  (rows >= 65 unused garbage)
  __bf16* sK    = sQ + 80 * 72;              // [144][72] (rows >= 129 unused garbage)
  __bf16* sVt   = sK + 144 * 72;             // [64][168] (V^T; m in [129,161) zeroed)
  float*  sS    = (float*)(sVt + 64 * 168);  // [80][160] scores -> probs
  float*  sBias = sS + 80 * 160;             // [80][160] (rows >= 65 unused garbage)

  const int tid  = threadIdx.x;
  const int wave = tid >> 5, lane = tid & 31;
  const int colL = lane & 15, half = lane >> 4;
  const int kbase8 = half * 8, kb16 = half * 16, rbase = half * 8;
  const int b = blockIdx.x, h = blockIdx.y;
  const int nWv = *nW_ptr;

  // window class -> fused bias+mask slice (uniform per block)
  int w = (nWv > 0) ? (b % nWv) : 0;
  int cls = 4;
  if (w == 0) cls = 0;
  else if (w == 1) cls = 1;
  if (nWv >= 2 && w == nWv - 2) cls = 2;
  else if (w == nWv - 1) cls = 3;
  const float* bsrc = biasTable + (((size_t)cls * NH + h) * NTOK) * 160;

  // ---- stage Q, K, bias slice (async direct-to-LDS), V^T (sync scatter) ----
  for (int idx = tid; idx < NTOK * 8; idx += 256) {
    int row = idx >> 3, seg = idx & 7;
    cp16B(Q + ((size_t)(b * NTOK + row)) * DIM + h * HD + seg * 8,
          sQ + row * 72 + seg * 8);
  }
  for (int idx = tid; idx < NTOK * 40; idx += 256) {  // 65 rows x 40 x 16B of bias
    int row = idx / 40, seg = idx % 40;
    cp16B(bsrc + row * 160 + seg * 4, sBias + row * 160 + seg * 4);
  }
  for (int idx = tid; idx < MTOK * 8; idx += 256) {
    int row = idx >> 3, seg = idx & 7;
    const size_t rb = (size_t)(b * MTOK + row) * (2 * DIM);
    cp16B(KV + rb + h * HD + seg * 8, sK + row * 72 + seg * 8);
    uint4 vv = *(const uint4*)(KV + rb + DIM + h * HD + seg * 8);
    const __bf16* e = (const __bf16*)&vv;
#pragma unroll
    for (int i = 0; i < 8; i++) sVt[(seg * 8 + i) * 168 + row] = e[i];
  }
  // zero V padding (m in [129,161)) so padded probs (=0) never multiply garbage
  for (int idx = tid; idx < 64 * 32; idx += 256) {
    int d = idx >> 5, mm = 129 + (idx & 31);
    sVt[d * 168 + mm] = (__bf16)0.0f;
  }
  wait_async_all();
  __syncthreads();

  // ---- S = (1/8) * Q K^T + (bias|mask) : branch-free epilogue ----
  for (int j = wave; j < 45; j += 8) {   // 5 M-tiles x 9 N-tiles
    int mt = j / 9, nt = j % 9;
    v8f acc = {};
#pragma unroll
    for (int s = 0; s < 2; s++) {        // K = 64 -> two K=32 steps
      v16bf a  = ld_fragA(sQ + (mt * 16 + colL) * 72 + s * 32, kbase8);
      v16bf bb = ld_frag32B(sK + (nt * 16 + colL) * 72 + s * 32 + kb16);
      acc = __builtin_amdgcn_wmma_f32_16x16x32_bf16(
          false, a, false, bb, (short)0, acc, false, false);
    }
#pragma unroll
    for (int r = 0; r < 8; r++) {
      int row = mt * 16 + rbase + r;
      int col = nt * 16 + colL;
      // cols >= 129 get NEG_BIG from the table; garbage there is overwritten by softmax
      sS[row * 160 + col] = acc[r] * 0.125f + sBias[row * 160 + col];
    }
  }
  __syncthreads();

  // ---- softmax over m (rows 0..64 only) ----
  if (tid < NTOK) {
    float* rowp = sS + tid * 160;
    float mx = NEG_BIG;
    for (int jj = 0; jj < MTOK; jj++) mx = fmaxf(mx, rowp[jj]);
    float sum = 0.f;
    for (int jj = 0; jj < MTOK; jj++) {
      float e = __expf(rowp[jj] - mx);
      rowp[jj] = e;
      sum += e;
    }
    float inv = 1.0f / sum;
    for (int jj = 0; jj < MTOK; jj++) rowp[jj] *= inv;
    for (int jj = MTOK; jj < 160; jj++) rowp[jj] = 0.f;  // padded K cols for P@V
  }
  __syncthreads();

  // ---- O = P @ V ----
  for (int j = wave; j < 20; j += 8) {   // 5 M-tiles x 4 D-tiles
    int mt = j / 4, nt = j % 4;
    v8f acc = {};
#pragma unroll
    for (int ks = 0; ks < 5; ks++) {     // K = 160 (129 valid + zero pad)
      const float* base = sS + (mt * 16 + colL) * 160 + ks * 32 + kb16;
      v16bf a;
#pragma unroll
      for (int i = 0; i < 8; i++) {
        a[i]     = (__bf16)base[i];
        a[8 + i] = (__bf16)base[8 + i];
      }
      v16bf bb = ld_frag32B(sVt + (nt * 16 + colL) * 168 + ks * 32 + kb16);
      acc = __builtin_amdgcn_wmma_f32_16x16x32_bf16(
          false, a, false, bb, (short)0, acc, false, false);
    }
#pragma unroll
    for (int r = 0; r < 8; r++) {
      int row = mt * 16 + rbase + r;
      if (row < NTOK)
        O[((size_t)(b * NTOK + row)) * DIM + h * HD + nt * 16 + colL] = (__bf16)acc[r];
    }
  }
}

// ---------------- host launch ----------------
extern "C" void kernel_launch(void* const* d_in, const int* in_sizes, int n_in,
                              void* d_out, int out_size, void* d_ws, size_t ws_size,
                              hipStream_t stream) {
  (void)in_sizes; (void)n_in; (void)out_size; (void)ws_size;
  const float* x   = (const float*)d_in[0];
  const float* x_  = (const float*)d_in[1];
  const unsigned char* ml = (const unsigned char*)d_in[2];
  const unsigned char* mr = (const unsigned char*)d_in[3];
  const int*   nW  = (const int*)d_in[4];
  const float* Wq  = (const float*)d_in[5];
  const float* bq  = (const float*)d_in[6];
  const float* Wkv = (const float*)d_in[7];
  const float* bkv = (const float*)d_in[8];
  const float* Wp  = (const float*)d_in[9];
  const float* bp  = (const float*)d_in[10];
  const float* rel = (const float*)d_in[11];
  const float* cs  = (const float*)d_in[12];
  const float* cu  = (const float*)d_in[13];
  const float* cd  = (const float*)d_in[14];
  float* out = (float*)d_out;

  const size_t NX   = (size_t)BNW * NTOK * DIM;        // 34,078,720
  const size_t NX_  = (size_t)BNW * MTOK * DIM;        // 67,633,152
  const size_t NWQ  = (size_t)DIM * DIM;
  const size_t NWKV = (size_t)DIM * 2 * DIM;
  const size_t NKV  = (size_t)BNW * MTOK * 2 * DIM;
  const size_t NBIAS = (size_t)5 * NH * NTOK * 160;    // 832,000 f32

  char* ws = (char*)d_ws;
  size_t off = 0;
  auto takeB = [&](size_t bytes) -> char* {
    off = (off + 255) & ~(size_t)255;
    char* p = ws + off;
    off += bytes;
    return p;
  };
  __bf16* xb   = (__bf16*)takeB(NX * 2);
  __bf16* x_b  = (__bf16*)takeB(NX_ * 2);
  __bf16* Wqb  = (__bf16*)takeB(NWQ * 2);
  __bf16* Wkvb = (__bf16*)takeB(NWKV * 2);
  __bf16* Wpb  = (__bf16*)takeB(NWQ * 2);
  __bf16* Qb   = (__bf16*)takeB(NX * 2);
  __bf16* KVb  = (__bf16*)takeB(NKV * 2);
  __bf16* Ob   = (__bf16*)takeB(NX * 2);
  float*  biasT = (float*)takeB(NBIAS * 4);

  auto cvtN = [&](const float* s, __bf16* d, size_t n) {
    unsigned blocks = (unsigned)((n / 4 + 255) / 256);
    cvt_f32_bf16<<<dim3(blocks), dim3(256), 0, stream>>>(s, d, (long long)n);
  };
  cvtN(x, xb, NX);
  cvtN(x_, x_b, NX_);
  cvtN(Wq, Wqb, NWQ);
  cvtN(Wkv, Wkvb, NWKV);
  cvtN(Wp, Wpb, NWQ);

  build_bias<<<dim3((unsigned)((NBIAS + 255) / 256)), dim3(256), 0, stream>>>(
      rel, cs, cu, cd, ml, mr, biasT);

  // Q = x @ Wq + bq      (M=33280, N=1024, K=1024)
  gemm_bf16<__bf16><<<dim3(33280 / 128, 1024 / 128), dim3(256), 0, stream>>>(
      xb, Wqb, bq, Qb, 33280, 1024, 1024);
  // KV = x_ @ Wkv + bkv  (M=66048, N=2048, K=1024)
  gemm_bf16<__bf16><<<dim3(66048 / 128, 2048 / 128), dim3(256), 0, stream>>>(
      x_b, Wkvb, bkv, KVb, 66048, 2048, 1024);

  const size_t smem = ((size_t)(80 * 72 + 144 * 72 + 64 * 168)) * 2 +
                      (size_t)80 * 160 * 4 * 2;   // + sS + sBias
  (void)hipFuncSetAttribute((const void*)attn_kernel,
                            hipFuncAttributeMaxDynamicSharedMemorySize, (int)smem);
  attn_kernel<<<dim3(BNW, NH), dim3(256), smem, stream>>>(Qb, KVb, biasT, nW, Ob);

  // out = O @ Wp + bp    (M=33280, N=1024, K=1024), f32 output
  gemm_bf16<float><<<dim3(33280 / 128, 1024 / 128), dim3(256), 0, stream>>>(
      Ob, Wpb, bp, out, 33280, 1024, 1024);
}
